// LSTMLayer_75058848465585
// MI455X (gfx1250) — compile-verified, matrix operand order checked by reading, not run on hardware
//
#include <hip/hip_runtime.h>
#include <hip/hip_bf16.h>
#include <cstdint>
#include <cstddef>

// ---------------------------------------------------------------------------
// Bidirectional 2-layer LSTM for MI455X (gfx1250), wave32 + WMMA bf16.
// B=32, T=512, E=512, H=512.  d_out = [avg_out (32*1024) | out (32*512*1024)]
// ---------------------------------------------------------------------------

typedef __bf16 bf16_t;
typedef __attribute__((ext_vector_type(16))) __bf16 v16bf;
typedef __attribute__((ext_vector_type(8)))  float  v8f;

#define B_SZ 32
#define T_SZ 512
#define E_SZ 512
#define H_SZ 512
#define G4   2048               // 4*H (gate columns)
#define M_ROWS (B_SZ * T_SZ)    // 16384 flattened (b,t) rows

// ---- bf16 helpers ---------------------------------------------------------
__device__ inline bf16_t f2bf(float f) {
  union { float f; uint32_t u; } in; in.f = f;
  uint32_t u = in.u;
  uint32_t r = (u + 0x7FFFu + ((u >> 16) & 1u)) >> 16;  // round-nearest-even
  union { uint16_t u; bf16_t b; } out; out.u = (uint16_t)r;
  return out.b;
}

// ---- WMMA 16-bit operand fragment load ------------------------------------
// CDNA5 16-bit A/B fragment (16x32 tile): lane L holds row (L&15); lanes 0-15
// carry K = 0..7 and 16..23, lanes 16-31 carry K = 8..15 and 24..31; each
// 8-element K-run is contiguous in memory -> two 16-byte vector loads.
union FragU { v16bf v; uint4 q[2]; };

__device__ inline v16bf load_frag_rows(const bf16_t* __restrict__ base,
                                       int stride, int lane) {
  const bf16_t* p = base + (size_t)(lane & 15) * stride + ((lane >> 4) << 3);
  FragU f;
  f.q[0] = *(const uint4*)(p);        // K = kbase + 0..7
  f.q[1] = *(const uint4*)(p + 16);   // K = kbase + 16..23
  return f.v;
}

__device__ inline float sigm(float x) { return 1.0f / (1.0f + __expf(-x)); }

// ---- grid-wide split barrier (generation counting, zeroed before launch) --
__device__ inline void grid_barrier(int* bar, int nwg, int gen) {
  __syncthreads();
  if (threadIdx.x == 0) {
    __threadfence();                 // release h stores
    atomicAdd(bar, 1);
    const int target = nwg * gen;
    while (__hip_atomic_load(bar, __ATOMIC_RELAXED, __HIP_MEMORY_SCOPE_AGENT) < target) {
      __builtin_amdgcn_s_sleep(2);
    }
    __threadfence();                 // acquire peers' h stores
  }
  __syncthreads();
}

// ---------------------------------------------------------------------------
// Kernels
// ---------------------------------------------------------------------------
__global__ void k_f32_to_bf16(const float* __restrict__ src,
                              bf16_t* __restrict__ dst, size_t n) {
  size_t i = (size_t)blockIdx.x * blockDim.x + threadIdx.x;
  if (i < n) dst[i] = f2bf(src[i]);
}

__global__ void k_zero_ints(int* __restrict__ p, int n) {
  int i = blockIdx.x * blockDim.x + threadIdx.x;
  if (i < n) p[i] = 0;
}

// xg[m,n] = sum_k X[m,k] * W[n,k] + bias[n]     (X: MxK, W: NxK, row-major)
// Register tile: each wave computes 32 rows x 64 cols = 2x4 WMMA tiles,
// reusing 2 A-fragments across 4 N-tiles and 4 B-fragments across 2 M-tiles
// (12 b128 loads per 8 WMMAs).
__global__ void __launch_bounds__(256)
k_gemm_xwT_bias(const bf16_t* __restrict__ X, const bf16_t* __restrict__ W,
                const float* __restrict__ bias, float* __restrict__ out,
                int M, int N, int K) {
  const int lane = threadIdx.x & 31;
  const int wave = threadIdx.x >> 5;
  const int m0 = (blockIdx.y * 8 + wave) * 32;   // 32-row strip per wave
  const int n0 = blockIdx.x * 64;                // 64-column strip

  v8f acc[2][4];
  #pragma unroll
  for (int i = 0; i < 2; ++i)
    #pragma unroll
    for (int t = 0; t < 4; ++t) acc[i][t] = (v8f){};

  for (int k = 0; k < K; k += 32) {
    v16bf a0 = load_frag_rows(X + (size_t)(m0 +  0) * K + k, K, lane);
    v16bf a1 = load_frag_rows(X + (size_t)(m0 + 16) * K + k, K, lane);
    #pragma unroll
    for (int t = 0; t < 4; ++t) {
      v16bf b = load_frag_rows(W + (size_t)(n0 + 16 * t) * K + k, K, lane);
      acc[0][t] = __builtin_amdgcn_wmma_f32_16x16x32_bf16(false, a0, false, b, (short)0, acc[0][t], false, false);
      acc[1][t] = __builtin_amdgcn_wmma_f32_16x16x32_bf16(false, a1, false, b, (short)0, acc[1][t], false, false);
    }
  }
  // C/D layout: lane&15 = column; VGPR v = row v (+8 for lanes 16-31)
  const int col = lane & 15;
  const int mh  = (lane >> 4) * 8;
  #pragma unroll
  for (int t = 0; t < 4; ++t) {
    const int n = n0 + 16 * t + col;
    const float bv = bias[n];
    #pragma unroll
    for (int i = 0; i < 2; ++i) {
      #pragma unroll
      for (int v = 0; v < 8; ++v) {
        out[(size_t)(m0 + 16 * i + mh + v) * N + n] = acc[i][t][v] + bv;
      }
    }
  }
}

// Persistent recurrence: 64 WGs = {2 dirs} x {32 blocks of 16 h-columns}.
// Per step: each of 8 waves computes one 16x16 gate tile (gate = wave>>1,
// mtile = wave&1) via bf16 WMMA over K=512, adds precomputed xg, then the WG
// applies i/f/g/o nonlinearity, updates c (LDS-resident) and h (global bf16
// double buffer, L2-resident), and crosses a grid barrier.
__global__ void __launch_bounds__(256)
k_lstm_recurrence(const float* __restrict__ xg_f, const float* __restrict__ xg_b,
                  const bf16_t* __restrict__ Whh_f, const bf16_t* __restrict__ Whh_b,
                  bf16_t* __restrict__ hbuf,   // [2 dirs][2 phases][32*512] bf16
                  bf16_t* __restrict__ xout,   // layer-0: bf16 [16384, 1024]
                  float*  __restrict__ fout,   // layer-1: f32  [16384, 1024]
                  int* __restrict__ bar, int last) {
  const int wg   = blockIdx.x;
  const int dir  = wg >> 5;        // 0 = forward, 1 = backward
  const int jj   = wg & 31;        // owns h columns [jj*16, jj*16+16)
  const int tid  = threadIdx.x;
  const int lane = tid & 31;
  const int wave = tid >> 5;
  const int gate = wave >> 1;      // 0:i 1:f 2:g 3:o
  const int mtile = wave & 1;      // batch rows 0-15 / 16-31
  const int nwg  = gridDim.x;

  const float*  xg  = dir ? xg_b  : xg_f;
  const bf16_t* Whh = dir ? Whh_b : Whh_f;
  bf16_t* hb0 = hbuf + (size_t)dir * 2 * B_SZ * H_SZ;
  bf16_t* hb1 = hb0 + (size_t)B_SZ * H_SZ;

  __shared__ float lds_g[4][B_SZ][16];   // staged gate pre-activations
  __shared__ float lds_c[B_SZ][16];      // persistent cell state

  // h(-1) = 0, c(-1) = 0 for our columns
  for (int e = tid; e < B_SZ * 16; e += 256) {
    const int m = e >> 4, c = e & 15;
    lds_c[m][c] = 0.0f;
    hb0[(size_t)m * H_SZ + jj * 16 + c] = f2bf(0.0f);
  }
  grid_barrier(bar, nwg, 1);

  const int gc0 = gate * H_SZ + jj * 16;  // gate-column base in [0, 2048)
  const int col = lane & 15;
  const int mh  = (lane >> 4) * 8;

  for (int s = 0; s < T_SZ; ++s) {
    const int t = dir ? (T_SZ - 1 - s) : s;
    const bf16_t* hprev = (s & 1) ? hb1 : hb0;
    bf16_t*       hnext = (s & 1) ? hb0 : hb1;

    // 16x16 gate tile: hprev[mtile*16.., :] @ Whh[gc0.., :]^T
    v8f acc = {};
    for (int k = 0; k < H_SZ; k += 32) {
      v16bf a = load_frag_rows(hprev + (size_t)(mtile * 16) * H_SZ + k, H_SZ, lane);
      v16bf b = load_frag_rows(Whh + (size_t)gc0 * H_SZ + k, H_SZ, lane);
      acc = __builtin_amdgcn_wmma_f32_16x16x32_bf16(false, a, false, b, (short)0, acc, false, false);
    }
    #pragma unroll
    for (int v = 0; v < 8; ++v) {
      const int bb = mtile * 16 + mh + v;
      const float xv = xg[((size_t)bb * T_SZ + t) * G4 + gc0 + col];
      lds_g[gate][bb][col] = acc[v] + xv;
    }
    __syncthreads();

    // elementwise LSTM cell update for our 32x16 h block
    for (int e = tid; e < B_SZ * 16; e += 256) {
      const int m = e >> 4, c = e & 15;
      const float gi = sigm(lds_g[0][m][c]);
      const float gf = sigm(lds_g[1][m][c]);
      const float gg = tanhf(lds_g[2][m][c]);
      const float go = sigm(lds_g[3][m][c]);
      const float cs = gf * lds_c[m][c] + gi * gg;
      const float h  = go * tanhf(cs);
      lds_c[m][c] = cs;
      hnext[(size_t)m * H_SZ + jj * 16 + c] = f2bf(h);
      const size_t row = (size_t)m * T_SZ + t;
      const size_t ch  = (size_t)dir * H_SZ + jj * 16 + c;
      if (last) fout[row * (size_t)(2 * H_SZ) + ch] = h;
      else      xout[row * (size_t)(2 * H_SZ) + ch] = f2bf(h);
    }
    grid_barrier(bar, nwg, s + 2);
  }
}

// avg_out[b,ch] = mean over t of out[b,t,ch]
__global__ void k_mean_over_t(const float* __restrict__ outseq,
                              float* __restrict__ avg) {
  const int i = blockIdx.x * blockDim.x + threadIdx.x;   // [0, 32*1024)
  const int b = i >> 10, ch = i & 1023;
  const float* p = outseq + (size_t)b * T_SZ * 1024 + ch;
  float s = 0.0f;
  for (int t = 0; t < T_SZ; ++t) s += p[(size_t)t * 1024];
  avg[i] = s * (1.0f / (float)T_SZ);
}

// ---------------------------------------------------------------------------
extern "C" void kernel_launch(void* const* d_in, const int* in_sizes, int n_in,
                              void* d_out, int out_size, void* d_ws, size_t ws_size,
                              hipStream_t stream) {
  (void)in_sizes; (void)n_in; (void)out_size; (void)ws_size;
  const float* word_emb = (const float*)d_in[0];
  const float* Wih0f = (const float*)d_in[1];
  const float* Whh0f = (const float*)d_in[2];
  const float* b0f   = (const float*)d_in[3];
  const float* Wih0b = (const float*)d_in[4];
  const float* Whh0b = (const float*)d_in[5];
  const float* b0b   = (const float*)d_in[6];
  const float* Wih1f = (const float*)d_in[7];
  const float* Whh1f = (const float*)d_in[8];
  const float* b1f   = (const float*)d_in[9];
  const float* Wih1b = (const float*)d_in[10];
  const float* Whh1b = (const float*)d_in[11];
  const float* b1b   = (const float*)d_in[12];

  // workspace carve-up
  char* ws = (char*)d_ws;
  size_t off = 0;
  auto carve = [&](size_t bytes) -> char* {
    char* p = ws + off;
    off += (bytes + 255) & ~(size_t)255;
    return p;
  };
  bf16_t* X0    = (bf16_t*)carve((size_t)M_ROWS * E_SZ * 2);        // 16 MB
  bf16_t* X1    = (bf16_t*)carve((size_t)M_ROWS * 2 * H_SZ * 2);    // 32 MB
  bf16_t* wih0f = (bf16_t*)carve((size_t)G4 * E_SZ * 2);
  bf16_t* wih0b = (bf16_t*)carve((size_t)G4 * E_SZ * 2);
  bf16_t* whh0f = (bf16_t*)carve((size_t)G4 * H_SZ * 2);
  bf16_t* whh0b = (bf16_t*)carve((size_t)G4 * H_SZ * 2);
  bf16_t* wih1f = (bf16_t*)carve((size_t)G4 * 2 * H_SZ * 2);
  bf16_t* wih1b = (bf16_t*)carve((size_t)G4 * 2 * H_SZ * 2);
  bf16_t* whh1f = (bf16_t*)carve((size_t)G4 * H_SZ * 2);
  bf16_t* whh1b = (bf16_t*)carve((size_t)G4 * H_SZ * 2);
  float*  xgF   = (float*)carve((size_t)M_ROWS * G4 * 4);           // 128 MB
  float*  xgB   = (float*)carve((size_t)M_ROWS * G4 * 4);           // 128 MB
  bf16_t* hbuf  = (bf16_t*)carve((size_t)2 * 2 * B_SZ * H_SZ * 2);  // 128 KB
  int*    bar   = (int*)carve(256);

  auto conv = [&](const float* s, bf16_t* d, size_t n) {
    const int blocks = (int)((n + 255) / 256);
    k_f32_to_bf16<<<dim3(blocks), dim3(256), 0, stream>>>(s, d, n);
  };
  conv(word_emb, X0, (size_t)M_ROWS * E_SZ);
  conv(Wih0f, wih0f, (size_t)G4 * E_SZ);
  conv(Wih0b, wih0b, (size_t)G4 * E_SZ);
  conv(Whh0f, whh0f, (size_t)G4 * H_SZ);
  conv(Whh0b, whh0b, (size_t)G4 * H_SZ);
  conv(Wih1f, wih1f, (size_t)G4 * 2 * H_SZ);
  conv(Wih1b, wih1b, (size_t)G4 * 2 * H_SZ);
  conv(Whh1f, whh1f, (size_t)G4 * H_SZ);
  conv(Whh1b, whh1b, (size_t)G4 * H_SZ);

  const dim3 blk(256);
  const dim3 gemm_grid(G4 / 64, M_ROWS / 256);  // (32, 64): 8 waves x 32 rows

  // ---- layer 0 ----
  k_gemm_xwT_bias<<<gemm_grid, blk, 0, stream>>>(X0, wih0f, b0f, xgF, M_ROWS, G4, E_SZ);
  k_gemm_xwT_bias<<<gemm_grid, blk, 0, stream>>>(X0, wih0b, b0b, xgB, M_ROWS, G4, E_SZ);
  k_zero_ints<<<1, 32, 0, stream>>>(bar, 32);
  k_lstm_recurrence<<<64, 256, 0, stream>>>(xgF, xgB, whh0f, whh0b, hbuf,
                                            X1, (float*)nullptr, bar, 0);

  // ---- layer 1 ----
  k_gemm_xwT_bias<<<gemm_grid, blk, 0, stream>>>(X1, wih1f, b1f, xgF, M_ROWS, G4, 2 * H_SZ);
  k_gemm_xwT_bias<<<gemm_grid, blk, 0, stream>>>(X1, wih1b, b1b, xgB, M_ROWS, G4, 2 * H_SZ);
  k_zero_ints<<<1, 32, 0, stream>>>(bar, 32);
  float* fout = (float*)d_out + B_SZ * 2 * H_SZ;  // out part after avg_out
  k_lstm_recurrence<<<64, 256, 0, stream>>>(xgF, xgB, whh1f, whh1b, hbuf,
                                            (bf16_t*)nullptr, fout, bar, 1);

  // ---- mean over time ----
  k_mean_over_t<<<(B_SZ * 2 * H_SZ) / 256, 256, 0, stream>>>(fout, (float*)d_out);
}
